// ModalityFusion_59554016526462
// MI455X (gfx1250) — compile-verified, matrix operand order checked by reading, not run on hardware
//
#include <hip/hip_runtime.h>
#include <hip/hip_bf16.h>

// ---------------------------------------------------------------------------
// MI455X / gfx1250 fused ModalityFusion.
// bf16 WMMA GEMMs (v_wmma_f32_16x16x32_bf16) with fragment-order LDS layout:
//  - B tile stored pre-swizzled so each lane's 16-half fragment is contiguous
//  - all 8 B fragments loaded (16x ds_load_b128) BEFORE the WMMA burst so the
//    compiler emits staggered s_wait_dscnt instead of wait-0 per WMMA
//  - A tile row-major with 64B rows: 2x aligned ds_load_b128 per fragment
//  - K-loop software-pipelined: next tile's global loads overlap the WMMAs
// Epilogue-fused scoring avoids materializing [S*B, 2404] hidden activations.
// ---------------------------------------------------------------------------

typedef __attribute__((ext_vector_type(16))) __bf16 v16bf;
typedef __attribute__((ext_vector_type(8)))  __bf16 v8bf;
typedef __attribute__((ext_vector_type(8)))  float  v8f;

#define BM 128
#define BN 128
#define BK 32
#define NT 8            // BN / 16 WMMA column subtiles per wave
#define BATCH 256

static __device__ __forceinline__ v8f wmma_bf16(v16bf a, v16bf b, v8f c) {
  return __builtin_amdgcn_wmma_f32_16x16x32_bf16(
      /*neg_a=*/false, a, /*neg_b=*/false, b,
      /*c_mod=*/(short)0, c, /*reuse_a=*/false, /*reuse_b=*/false);
}

static __device__ __forceinline__ v16bf cat8(v8bf lo, v8bf hi) {
  return __builtin_shufflevector(lo, hi, 0, 1, 2, 3, 4, 5, 6, 7,
                                 8, 9, 10, 11, 12, 13, 14, 15);
}

// ---- per-thread register staging -----------------------------------------
// A: thread t covers row r = t/2, cols c..c+15 (c = (t&1)*16) of the A tile.
static __device__ __forceinline__ void loadA(const float* __restrict__ A,
                                             int M, int K, int m0, int k0,
                                             int tid, float r[16]) {
  const int row = m0 + (tid >> 1);
  const int c   = (tid & 1) * 16;
  if (row < M && (k0 + BK) <= K) {
    const float* src = A + (long)row * K + k0 + c;
#pragma unroll
    for (int i = 0; i < 16; i += 4) {
      float4 v = *(const float4*)(src + i);
      r[i + 0] = v.x; r[i + 1] = v.y; r[i + 2] = v.z; r[i + 3] = v.w;
    }
  } else {
#pragma unroll
    for (int i = 0; i < 16; ++i) {
      int k = k0 + c + i;
      r[i] = (row < M && k < K) ? A[(long)row * K + k] : 0.0f;
    }
  }
}

// B: thread t gathers fragment t (nt = t>>5, lane = t&31): 16 halves of the
// column col = nt*16 + (lane&15), K = kb..kb+7 then kb+16..kb+23, kb=(lane>>4)*8.
static __device__ __forceinline__ void loadB(const float* __restrict__ Bw,
                                             int K, int N, int n0, int k0,
                                             int tid, float r[16]) {
  const int ln  = tid & 31;
  const int nt  = tid >> 5;
  const int col = n0 + nt * 16 + (ln & 15);
  const int kb  = (ln >> 4) * 8;
#pragma unroll
  for (int h = 0; h < 2; ++h) {
#pragma unroll
    for (int i = 0; i < 8; ++i) {
      int gk = k0 + kb + h * 16 + i;
      r[h * 8 + i] = (gk < K && col < N) ? Bw[(long)gk * N + col] : 0.0f;
    }
  }
}

// MODE 0: fused score:  atomicAdd(score[row], sum_n tanh(acc + fp[b,n]) * w2[n])
// MODE 1: out[row,n] = tanh(acc + bias[n])
// MODE 2: out[row,n] = acc + bias[n]
template <int MODE>
__global__ __launch_bounds__(256)
void gemm_wmma(const float* __restrict__ A,   // [M, K] row-major f32
               const float* __restrict__ Bw,  // [K, N] row-major f32
               const float* __restrict__ bias,// [N]      (MODE 1/2)
               const float* __restrict__ fp,  // [BATCH,N](MODE 0, includes b1)
               const float* __restrict__ w2,  // [N]      (MODE 0)
               float* __restrict__ out,       // [M] (MODE0) or [M,N]
               int M, int K, int N, int Kpad)
{
  __shared__ __bf16 As[BM][BK];          // 8 KB, 64B rows (16B-aligned chunks)
  __shared__ __bf16 BsF[NT * 32 * 16];   // 8 KB, fragment-order: [nt][lane][16]

  const int m0   = blockIdx.x * BM;
  const int n0   = blockIdx.y * BN;
  const int tid  = threadIdx.x;
  const int lane = tid & 31;
  const int wave = tid >> 5;             // 8 waves, each owns a 16-row strip
  const int hf   = lane >> 4;
  const int l15  = lane & 15;
  const int kb   = hf * 8;

  v8f acc[NT];
#pragma unroll
  for (int i = 0; i < NT; ++i)
#pragma unroll
    for (int j = 0; j < 8; ++j) acc[i][j] = 0.0f;

  float ra[16], rb[16];
  loadA(A, M, K, m0, 0, tid, ra);
  loadB(Bw, K, N, n0, 0, tid, rb);

  for (int k0 = 0; k0 < Kpad; k0 += BK) {
    __syncthreads();                      // previous iter's LDS reads done
    // ---- publish staged tile (f32 -> bf16, vector LDS stores) ----
    {
      v8bf pa0, pa1;
#pragma unroll
      for (int i = 0; i < 8; ++i) { pa0[i] = (__bf16)ra[i]; pa1[i] = (__bf16)ra[8 + i]; }
      __bf16* dst = &As[tid >> 1][(tid & 1) * 16];
      *(v8bf*)(dst)     = pa0;            // ds_store_b128
      *(v8bf*)(dst + 8) = pa1;            // ds_store_b128
      v8bf pb0, pb1;
#pragma unroll
      for (int i = 0; i < 8; ++i) { pb0[i] = (__bf16)rb[i]; pb1[i] = (__bf16)rb[8 + i]; }
      __bf16* bdst = &BsF[tid * 16];
      *(v8bf*)(bdst)     = pb0;           // ds_store_b128
      *(v8bf*)(bdst + 8) = pb1;           // ds_store_b128
    }
    __syncthreads();

    // ---- prefetch next tile into registers (overlaps the WMMA burst) ----
    if (k0 + BK < Kpad) {
      loadA(A, M, K, m0, k0 + BK, tid, ra);
      loadB(Bw, K, N, n0, k0 + BK, tid, rb);
    }

    // ---- fragments: issue ALL ds_load_b128 first, then the WMMA burst ----
    const __bf16* arow = &As[wave * 16 + l15][0];
    v16bf af = cat8(*(const v8bf*)(arow + kb), *(const v8bf*)(arow + kb + 16));
    v16bf bfrag[NT];
#pragma unroll
    for (int nt = 0; nt < NT; ++nt) {
      const __bf16* fb = &BsF[(nt * 32 + lane) * 16];
      bfrag[nt] = cat8(*(const v8bf*)(fb), *(const v8bf*)(fb + 8));
    }
#pragma unroll
    for (int nt = 0; nt < NT; ++nt)
      acc[nt] = wmma_bf16(af, bfrag[nt], acc[nt]);
  }

  // ---- epilogue: C layout row = base + vgpr + 8*half, col = l15 ----
  if constexpr (MODE == 0) {
#pragma unroll
    for (int v = 0; v < 8; ++v) {
      const int row = m0 + wave * 16 + hf * 8 + v;
      const int bb  = row & (BATCH - 1);       // row = s*BATCH + b
      float part = 0.0f;
#pragma unroll
      for (int nt = 0; nt < NT; ++nt) {
        int n = n0 + nt * 16 + l15;
        if (n < N)
          part += tanhf(acc[nt][v] + fp[(long)bb * N + n]) * w2[n];
      }
      // reduce the 16 column-lanes of this row half (xor masks stay in-half)
      part += __shfl_xor(part, 1, 32);
      part += __shfl_xor(part, 2, 32);
      part += __shfl_xor(part, 4, 32);
      part += __shfl_xor(part, 8, 32);
      if (l15 == 0 && row < M) atomicAdd(&out[row], part);
    }
  } else {
#pragma unroll
    for (int v = 0; v < 8; ++v) {
      const int row = m0 + wave * 16 + hf * 8 + v;
      if (row >= M) continue;
#pragma unroll
      for (int nt = 0; nt < NT; ++nt) {
        int n = n0 + nt * 16 + l15;
        if (n < N) {
          float x = acc[nt][v] + bias[n];
          out[(long)row * N + n] = (MODE == 1) ? tanhf(x) : x;
        }
      }
    }
  }
}

// score[i] = b2 (scalar from device memory)
__global__ void fill_from_scalar(float* p, int n, const float* __restrict__ s) {
  int i = blockIdx.x * blockDim.x + threadIdx.x;
  if (i < n) p[i] = s[0];
}

// per batch-column b: w[s,b] = sum_k softmax_s(score_k[s,b])
__global__ void softmax3_combine(const float* __restrict__ s0,
                                 const float* __restrict__ s1,
                                 const float* __restrict__ s2,
                                 float* __restrict__ w, int S) {
  const int b = threadIdx.x;                 // 256 threads, grid = 1
  const float* ss[3] = {s0, s1, s2};
  for (int s = 0; s < S; ++s) w[s * BATCH + b] = 0.0f;
  for (int k = 0; k < 3; ++k) {
    float mx = -3.0e38f;
    for (int s = 0; s < S; ++s) mx = fmaxf(mx, ss[k][s * BATCH + b]);
    float sum = 0.0f;
    for (int s = 0; s < S; ++s) sum += __expf(ss[k][s * BATCH + b] - mx);
    float inv = 1.0f / sum;
    for (int s = 0; s < S; ++s)
      w[s * BATCH + b] += __expf(ss[k][s * BATCH + b] - mx) * inv;
  }
}

// pooled[b,d] = (1/(3S)) * sum_s w[s,b] * seq[s,b,d]
__global__ void weighted_pool(const float* __restrict__ seq,
                              const float* __restrict__ w,
                              float* __restrict__ out, int S, int D) {
  int idx = blockIdx.x * blockDim.x + threadIdx.x;
  if (idx >= BATCH * D) return;
  int b = idx / D, d = idx - b * D;
  float a = 0.0f;
  for (int s = 0; s < S; ++s)
    a += w[s * BATCH + b] * seq[((long)s * BATCH + b) * D + d];
  out[idx] = a / (3.0f * (float)S);
}

// s[b] = dot(H[b,0:512], w2) + b2
__global__ void dot512_bias(const float* __restrict__ H,
                            const float* __restrict__ w2,
                            const float* __restrict__ b2,
                            float* __restrict__ s) {
  __shared__ float red[256];
  const int b = blockIdx.x, t = threadIdx.x;
  float a = 0.0f;
  for (int n = t; n < 512; n += 256) a += H[(long)b * 512 + n] * w2[n];
  red[t] = a;
  __syncthreads();
  for (int off = 128; off > 0; off >>= 1) {
    if (t < off) red[t] += red[t + off];
    __syncthreads();
  }
  if (t == 0) s[b] = red[0] + b2[0];
}

// out[b,n] = softmax([si,st,sa])[.]  dot  [Vi,Vt,Va][b,n]
__global__ void final_combine(const float* __restrict__ si,
                              const float* __restrict__ st,
                              const float* __restrict__ sa,
                              const float* __restrict__ Vi,
                              const float* __restrict__ Vt,
                              const float* __restrict__ Va,
                              float* __restrict__ out) {
  int idx = blockIdx.x * blockDim.x + threadIdx.x;   // 256*512
  int b = idx >> 9;
  float a0 = si[b], a1 = st[b], a2 = sa[b];
  float mx = fmaxf(a0, fmaxf(a1, a2));
  float e0 = __expf(a0 - mx), e1 = __expf(a1 - mx), e2 = __expf(a2 - mx);
  float inv = 1.0f / (e0 + e1 + e2);
  out[idx] = (e0 * Vi[idx] + e1 * Vt[idx] + e2 * Va[idx]) * inv;
}

// ---------------------------------------------------------------------------
static inline int ceilDiv(int a, int b) { return (a + b - 1) / b; }
static inline int pad32(int x) { return (x + 31) & ~31; }

extern "C" void kernel_launch(void* const* d_in, const int* in_sizes, int n_in,
                              void* d_out, int out_size, void* d_ws, size_t ws_size,
                              hipStream_t stream)
{
  auto in = [&](int i) { return (const float*)d_in[i]; };
  const float* img_feat = in(0);   // [256,1024]
  const float* img_seq  = in(1);   // [196,256,1024]
  const float* txt_feat = in(2);   // [256,512]
  const float* txt_seq  = in(3);   // [75,256,512]
  const float* att_feat = in(4);   // [256,200]
  const float* att_seq  = in(5);   // [5,256,200]

  // jax pytree: dict keys sorted -> blocks 'aa','ia','mod','ta'; within att
  // blocks: W11,W12,W13,W21,W22,W23,b11,b12,b13,b21,b22,b23.
  const int AA = 6, IA = 18, MOD = 30, TA = 48;

  struct Br { const float* feat; int Dk; int d;
              const float* W1; const float* b1; const float* W2; const float* b2; };
  struct Mo { const float* seq; int S; int D1; Br br[3];
              const float *W1m, *b1m, *W2m, *b2m, *W3m, *b3m; };

  Mo mo[3];
  // image ('ia'): f1=img, f2=txt, f3=att ; d = 1024,768,612
  mo[0].seq = img_seq; mo[0].S = 196; mo[0].D1 = 1024;
  mo[0].br[0] = { img_feat, 1024, 1024, in(IA+0), in(IA+6), in(IA+3), in(IA+9)  };
  mo[0].br[1] = { txt_feat,  512,  768, in(IA+1), in(IA+7), in(IA+4), in(IA+10) };
  mo[0].br[2] = { att_feat,  200,  612, in(IA+2), in(IA+8), in(IA+5), in(IA+11) };
  mo[0].W1m = in(MOD+6); mo[0].b1m = in(MOD+9);  mo[0].W2m = in(MOD+7);
  mo[0].b2m = in(MOD+10); mo[0].W3m = in(MOD+8); mo[0].b3m = in(MOD+11);
  // text ('ta'): d = 512,768,356
  mo[1].seq = txt_seq; mo[1].S = 75; mo[1].D1 = 512;
  mo[1].br[0] = { txt_feat,  512,  512, in(TA+0), in(TA+6), in(TA+3), in(TA+9)  };
  mo[1].br[1] = { img_feat, 1024,  768, in(TA+1), in(TA+7), in(TA+4), in(TA+10) };
  mo[1].br[2] = { att_feat,  200,  356, in(TA+2), in(TA+8), in(TA+5), in(TA+11) };
  mo[1].W1m = in(MOD+12); mo[1].b1m = in(MOD+15); mo[1].W2m = in(MOD+13);
  mo[1].b2m = in(MOD+16); mo[1].W3m = in(MOD+14); mo[1].b3m = in(MOD+17);
  // attribute ('aa'): d = 200,612,356
  mo[2].seq = att_seq; mo[2].S = 5; mo[2].D1 = 200;
  mo[2].br[0] = { att_feat,  200,  200, in(AA+0), in(AA+6), in(AA+3), in(AA+9)  };
  mo[2].br[1] = { img_feat, 1024,  612, in(AA+1), in(AA+7), in(AA+4), in(AA+10) };
  mo[2].br[2] = { txt_feat,  512,  356, in(AA+2), in(AA+8), in(AA+5), in(AA+11) };
  mo[2].W1m = in(MOD+0); mo[2].b1m = in(MOD+3); mo[2].W2m = in(MOD+1);
  mo[2].b2m = in(MOD+4); mo[2].W3m = in(MOD+2); mo[2].b3m = in(MOD+5);

  // ---- workspace bump allocator ----
  char* wsb = (char*)d_ws;
  size_t off = 0;
  auto alloc = [&](size_t bytes) -> float* {
    float* p = (float*)(wsb + off);
    off = (off + bytes + 255) & ~(size_t)255;
    return p;
  };
  float *fp[3][3], *sc[3][3], *wcomb[3], *pooled[3], *H[3], *V[3], *sm[3];
  for (int m = 0; m < 3; ++m) {
    for (int k = 0; k < 3; ++k) {
      fp[m][k] = alloc((size_t)BATCH * mo[m].br[k].d * 4);
      sc[m][k] = alloc((size_t)mo[m].S * BATCH * 4);
    }
    wcomb[m]  = alloc((size_t)mo[m].S * BATCH * 4);
    pooled[m] = alloc((size_t)BATCH * mo[m].D1 * 4);
    H[m]      = alloc((size_t)BATCH * 512 * 4);
    V[m]      = alloc((size_t)BATCH * 512 * 4);
    sm[m]     = alloc((size_t)BATCH * 4);
  }

  // ---- phase 1: per-branch feature projections + fused score GEMMs ----
  for (int m = 0; m < 3; ++m) {
    const Mo& M_ = mo[m];
    for (int k = 0; k < 3; ++k) {
      const Br& br = M_.br[k];
      // fp = feat @ W1[D1:, :] + b1   (MODE 2)
      dim3 g1(ceilDiv(BATCH, BM), ceilDiv(br.d, BN));
      gemm_wmma<2><<<g1, 256, 0, stream>>>(
          br.feat, br.W1 + (long)M_.D1 * br.d, br.b1, nullptr, nullptr,
          fp[m][k], BATCH, br.Dk, br.d, pad32(br.Dk));
      // score init with b2
      int nsc = M_.S * BATCH;
      fill_from_scalar<<<ceilDiv(nsc, 256), 256, 0, stream>>>(sc[m][k], nsc, br.b2);
      // score += sum_n tanh(seq@W1_top + fp) * W2   (MODE 0, fused epilogue)
      dim3 g2(ceilDiv(M_.S * BATCH, BM), ceilDiv(br.d, BN));
      gemm_wmma<0><<<g2, 256, 0, stream>>>(
          M_.seq, br.W1, nullptr, fp[m][k], br.W2,
          sc[m][k], M_.S * BATCH, M_.D1, br.d, pad32(M_.D1));
    }
  }

  // ---- phase 2: softmax over S, weighted pooling, modality nets ----
  for (int m = 0; m < 3; ++m) {
    const Mo& M_ = mo[m];
    softmax3_combine<<<1, 256, 0, stream>>>(sc[m][0], sc[m][1], sc[m][2],
                                            wcomb[m], M_.S);
    int tot = BATCH * M_.D1;
    weighted_pool<<<ceilDiv(tot, 256), 256, 0, stream>>>(
        M_.seq, wcomb[m], pooled[m], M_.S, M_.D1);

    dim3 gh(ceilDiv(BATCH, BM), ceilDiv(512, BN));
    gemm_wmma<1><<<gh, 256, 0, stream>>>(
        pooled[m], M_.W1m, M_.b1m, nullptr, nullptr, H[m],
        BATCH, M_.D1, 512, pad32(M_.D1));
    dot512_bias<<<BATCH, 256, 0, stream>>>(H[m], M_.W2m, M_.b2m, sm[m]);
    gemm_wmma<1><<<gh, 256, 0, stream>>>(
        pooled[m], M_.W3m, M_.b3m, nullptr, nullptr, V[m],
        BATCH, M_.D1, 512, pad32(M_.D1));
  }

  // ---- phase 3: 3-way modality softmax combine -> [256,512] ----
  final_combine<<<(BATCH * 512) / 256, 256, 0, stream>>>(
      sm[0], sm[1], sm[2], V[0], V[1], V[2], (float*)d_out);
}
